// NoSharedRnnAgent_64647847739630
// MI455X (gfx1250) — matrix operand order repaired and checked by reading, not compiled, not run on hardware
//
#include <hip/hip_runtime.h>

// ---------------------------------------------------------------------------
// MI455X (gfx1250) multi-agent GRU forward.
// bf16 WMMA (v_wmma_f32_16x16x32_bf16) with fp32 accumulation; fp32 operands
// converted to bf16 once during LDS staging (shared across consuming waves).
// Fragment loads follow the documented dense 16-bit A interleave
// (K {0-7,16-23}/{8-15,24-31} per lane group) and the same interleave for B.
// GRU phase-0 A staging (bf16->bf16) uses GLOBAL_LOAD_ASYNC_TO_LDS_B128.
// ---------------------------------------------------------------------------

typedef __bf16 bf16_t;
typedef __attribute__((ext_vector_type(4)))  float  v4f;
typedef __attribute__((ext_vector_type(8)))  float  v8f;
typedef __attribute__((ext_vector_type(4)))  bf16_t v4bf;
typedef __attribute__((ext_vector_type(8)))  bf16_t v8bf;
typedef __attribute__((ext_vector_type(16))) bf16_t v16bf;
typedef __attribute__((ext_vector_type(4)))  int    v4i_t;

#define A_N   32
#define IN_N  512
#define H_N   512
#define NA_N  64
#define B_N   256
#define BA    (B_N * A_N)
#define QSZ   (BA * NA_N)
#define KT    32          // K slab per stage
#define SP    40          // padded LDS row stride in halves (80B: 16B-aligned frags)

#if __has_builtin(__builtin_amdgcn_global_load_async_to_lds_b128)
#define HAVE_ASYNC_LDS 1
#endif

// A operand: documented interleave. Lane group hs holds K = [8*hs, 8*hs+8) and
// [16+8*hs, 16+8*hs+8) of the 32-wide slab.
__device__ __forceinline__ v16bf ld_fragA(const bf16_t* row, int hs) {
  v8bf lo = *(const v8bf*)(row + hs * 8);
  v8bf hi = *(const v8bf*)(row + 16 + hs * 8);
  v16bf r;
#pragma unroll
  for (int i = 0; i < 8; ++i) { r[i] = lo[i]; r[i + 8] = hi[i]; }
  return r;
}

// B operand: same interleave as A. Lane group hs holds K = [8*hs, 8*hs+8) and
// [16+8*hs, 16+8*hs+8).
__device__ __forceinline__ v16bf ld_fragB(const bf16_t* row, int hs) {
  v8bf lo = *(const v8bf*)(row + hs * 8);
  v8bf hi = *(const v8bf*)(row + 16 + hs * 8);
  v16bf r;
#pragma unroll
  for (int i = 0; i < 8; ++i) { r[i] = lo[i]; r[i + 8] = hi[i]; }
  return r;
}

__device__ __forceinline__ v8f wmma_bf16(v16bf a, v16bf b, v8f c) {
  return __builtin_amdgcn_wmma_f32_16x16x32_bf16(false, a, false, b, (short)0, c, false, false);
}

__device__ __forceinline__ v4bf cvt4(v4f x) {
  v4bf h;
  h[0] = (bf16_t)x[0]; h[1] = (bf16_t)x[1];
  h[2] = (bf16_t)x[2]; h[3] = (bf16_t)x[3];
  return h;
}

__device__ __forceinline__ float sigmoid_f(float x) { return 1.0f / (1.0f + __expf(-x)); }
__device__ __forceinline__ float tanh_f(float x) {
  float e = __expf(-2.0f * x);
  return (1.0f - e) / (1.0f + e);
}

#define MMA4(ACC)                                        \
  do {                                                   \
    ACC[0][0] = wmma_bf16(af0, bf0, ACC[0][0]);          \
    ACC[0][1] = wmma_bf16(af0, bf1, ACC[0][1]);          \
    ACC[1][0] = wmma_bf16(af1, bf0, ACC[1][0]);          \
    ACC[1][1] = wmma_bf16(af1, bf1, ACC[1][1]);          \
  } while (0)

// ------------------------- kernel 1: x1 = relu(X W1^T + b1) -----------------
__global__ __launch_bounds__(256) void k_fc1(const float* __restrict__ X,
                                             const float* __restrict__ W1,
                                             const float* __restrict__ b1,
                                             bf16_t* __restrict__ x1ws) {
  __shared__ bf16_t Al[128 * SP];
  __shared__ bf16_t Bl[64 * SP];
  const int a  = blockIdx.x;
  const int Bb = blockIdx.y * 128;
  const int Nb = blockIdx.z * 64;
  const int t  = threadIdx.x;
  const int lane = t & 31, wave = t >> 5;
  const int wr = wave >> 1, wc = wave & 1;   // 4x2 waves, 32x32 tile each
  const int hs = lane >> 4, mr = lane & 15;

  v8f acc[2][2] = {};

  for (int k0 = 0; k0 < IN_N; k0 += KT) {
    // stage X tile: 128 rows x 32 k (fp32 -> bf16)
#pragma unroll
    for (int it = 0; it < 4; ++it) {
      int q = t + it * 256, r = q >> 3, kq = q & 7;
      v4f x = *(const v4f*)(X + (long)((Bb + r) * A_N + a) * IN_N + k0 + kq * 4);
      *(v4bf*)(Al + r * SP + kq * 4) = cvt4(x);
    }
    // stage W1 tile: 64 rows x 32 k
#pragma unroll
    for (int it = 0; it < 2; ++it) {
      int q = t + it * 256, r = q >> 3, kq = q & 7;
      v4f x = *(const v4f*)(W1 + (long)(a * H_N + Nb + r) * IN_N + k0 + kq * 4);
      *(v4bf*)(Bl + r * SP + kq * 4) = cvt4(x);
    }
    __syncthreads();
    const bf16_t* ar = Al + (wr * 32 + mr) * SP;
    const bf16_t* br = Bl + (wc * 32 + mr) * SP;
    v16bf af0 = ld_fragA(ar, hs);
    v16bf af1 = ld_fragA(ar + 16 * SP, hs);
    v16bf bf0 = ld_fragB(br, hs);
    v16bf bf1 = ld_fragB(br + 16 * SP, hs);
    MMA4(acc);
    __syncthreads();
  }

#pragma unroll
  for (int i = 0; i < 2; ++i)
#pragma unroll
    for (int j = 0; j < 2; ++j) {
      int n = Nb + wc * 32 + j * 16 + mr;
      float bias = b1[a * H_N + n];
#pragma unroll
      for (int v = 0; v < 8; ++v) {
        int b = Bb + wr * 32 + i * 16 + v + 8 * hs;
        float val = acc[i][j][v] + bias;
        val = fmaxf(val, 0.0f);
        x1ws[(long)(b * A_N + a) * H_N + n] = (bf16_t)val;
      }
    }
}

// ------------------------- kernel 2: fused GRU cell -------------------------
// acc_r/acc_z accumulate BOTH gx and gh contributions across two K phases.
__global__ __launch_bounds__(256) void k_gru(const bf16_t* __restrict__ x1ws,
                                             const float* __restrict__ Hin,
                                             const float* __restrict__ Wih,
                                             const float* __restrict__ bih,
                                             const float* __restrict__ Whh,
                                             const float* __restrict__ bhh,
                                             float* __restrict__ Hout) {
  __shared__ bf16_t Al[128 * SP];
  __shared__ bf16_t Bl[3 * 64 * SP];
  const int a  = blockIdx.x;
  const int Bb = blockIdx.y * 128;
  const int Nb = blockIdx.z * 64;
  const int t  = threadIdx.x;
  const int lane = t & 31, wave = t >> 5;
  const int wr = wave >> 1, wc = wave & 1;
  const int hs = lane >> 4, mr = lane & 15;

  v8f accR[2][2] = {}, accZ[2][2] = {}, accNX[2][2] = {}, accNH[2][2] = {};

#pragma unroll
  for (int phase = 0; phase < 2; ++phase) {
    const float* Wsrc = phase ? Whh : Wih;
    for (int k0 = 0; k0 < H_N; k0 += KT) {
      // stage A tile: phase0 = x1 (bf16, async DMA to LDS), phase1 = h_in (cvt)
      if (phase == 0) {
#if defined(HAVE_ASYNC_LDS)
#pragma unroll
        for (int it = 0; it < 2; ++it) {
          int c = t + it * 256, r = c >> 2, k8 = c & 3;   // 16B chunks
          const bf16_t* src = x1ws + (long)((Bb + r) * A_N + a) * H_N + k0 + k8 * 8;
          bf16_t* dst = Al + r * SP + k8 * 8;
          __builtin_amdgcn_global_load_async_to_lds_b128((v4i_t*)src, (v4i_t*)dst, 0, 0);
        }
        asm volatile("s_wait_asynccnt 0x0" ::: "memory");
#else
#pragma unroll
        for (int it = 0; it < 4; ++it) {
          int q = t + it * 256, r = q >> 3, kq = q & 7;
          *(v4bf*)(Al + r * SP + kq * 4) =
              *(const v4bf*)(x1ws + (long)((Bb + r) * A_N + a) * H_N + k0 + kq * 4);
        }
#endif
      } else {
#pragma unroll
        for (int it = 0; it < 4; ++it) {
          int q = t + it * 256, r = q >> 3, kq = q & 7;
          v4f x = *(const v4f*)(Hin + (long)((Bb + r) * A_N + a) * H_N + k0 + kq * 4);
          *(v4bf*)(Al + r * SP + kq * 4) = cvt4(x);
        }
      }
      // stage the 3 gate weight tiles: rows {n, n+H, n+2H} of W (64 x 32 each)
#pragma unroll
      for (int it = 0; it < 6; ++it) {
        int q = t + it * 256;
        int g = q >> 9, rem = q & 511, r = rem >> 3, kq = rem & 7;
        v4f x = *(const v4f*)(Wsrc + ((long)(a * 3 + g) * H_N + Nb + r) * H_N + k0 + kq * 4);
        *(v4bf*)(Bl + (g * 64 + r) * SP + kq * 4) = cvt4(x);
      }
      __syncthreads();
      const bf16_t* ar = Al + (wr * 32 + mr) * SP;
      v16bf af0 = ld_fragA(ar, hs);
      v16bf af1 = ld_fragA(ar + 16 * SP, hs);
#pragma unroll
      for (int g = 0; g < 3; ++g) {
        const bf16_t* br = Bl + (g * 64 + wc * 32 + mr) * SP;
        v16bf bf0 = ld_fragB(br, hs);
        v16bf bf1 = ld_fragB(br + 16 * SP, hs);
        if (g == 0)           MMA4(accR);
        else if (g == 1)      MMA4(accZ);
        else if (phase == 0)  MMA4(accNX);
        else                  MMA4(accNH);
      }
      __syncthreads();
    }
  }

  const float* bih_a = bih + a * 3 * H_N;
  const float* bhh_a = bhh + a * 3 * H_N;
#pragma unroll
  for (int i = 0; i < 2; ++i)
#pragma unroll
    for (int j = 0; j < 2; ++j) {
      int n = Nb + wc * 32 + j * 16 + mr;
      float br  = bih_a[n] + bhh_a[n];
      float bz  = bih_a[H_N + n] + bhh_a[H_N + n];
      float bnx = bih_a[2 * H_N + n];
      float bnh = bhh_a[2 * H_N + n];
#pragma unroll
      for (int v = 0; v < 8; ++v) {
        int b = Bb + wr * 32 + i * 16 + v + 8 * hs;
        long off = (long)(b * A_N + a) * H_N + n;
        float r  = sigmoid_f(accR[i][j][v] + br);
        float z  = sigmoid_f(accZ[i][j][v] + bz);
        float nn = tanh_f(accNX[i][j][v] + bnx + r * (accNH[i][j][v] + bnh));
        float hp = Hin[off];
        Hout[off] = (1.0f - z) * nn + z * hp;
      }
    }
}

// ------------------------- kernel 3: q = h' W2^T + b2 -----------------------
__global__ __launch_bounds__(256) void k_fc2(const float* __restrict__ Hst,
                                             const float* __restrict__ W2,
                                             const float* __restrict__ b2,
                                             float* __restrict__ Q) {
  __shared__ bf16_t Al[128 * SP];
  __shared__ bf16_t Bl[64 * SP];
  const int a  = blockIdx.x;
  const int Bb = blockIdx.y * 128;
  const int t  = threadIdx.x;
  const int lane = t & 31, wave = t >> 5;
  const int wr = wave >> 1, wc = wave & 1;   // 2 wave-cols cover all 64 outputs
  const int hs = lane >> 4, mr = lane & 15;

  v8f acc[2][2] = {};

  for (int k0 = 0; k0 < H_N; k0 += KT) {
#pragma unroll
    for (int it = 0; it < 4; ++it) {
      int q = t + it * 256, r = q >> 3, kq = q & 7;
      v4f x = *(const v4f*)(Hst + (long)((Bb + r) * A_N + a) * H_N + k0 + kq * 4);
      *(v4bf*)(Al + r * SP + kq * 4) = cvt4(x);
    }
#pragma unroll
    for (int it = 0; it < 2; ++it) {
      int q = t + it * 256, r = q >> 3, kq = q & 7;
      v4f x = *(const v4f*)(W2 + (long)(a * NA_N + r) * H_N + k0 + kq * 4);
      *(v4bf*)(Bl + r * SP + kq * 4) = cvt4(x);
    }
    __syncthreads();
    const bf16_t* ar = Al + (wr * 32 + mr) * SP;
    const bf16_t* br = Bl + (wc * 32 + mr) * SP;
    v16bf af0 = ld_fragA(ar, hs);
    v16bf af1 = ld_fragA(ar + 16 * SP, hs);
    v16bf bf0 = ld_fragB(br, hs);
    v16bf bf1 = ld_fragB(br + 16 * SP, hs);
    MMA4(acc);
    __syncthreads();
  }

#pragma unroll
  for (int i = 0; i < 2; ++i)
#pragma unroll
    for (int j = 0; j < 2; ++j) {
      int n = wc * 32 + j * 16 + mr;
      float bias = b2[a * NA_N + n];
#pragma unroll
      for (int v = 0; v < 8; ++v) {
        int b = Bb + wr * 32 + i * 16 + v + 8 * hs;
        Q[(long)(b * A_N + a) * NA_N + n] = acc[i][j][v] + bias;
      }
    }
}

// ---------------------------------------------------------------------------
extern "C" void kernel_launch(void* const* d_in, const int* in_sizes, int n_in,
                              void* d_out, int out_size, void* d_ws, size_t ws_size,
                              hipStream_t stream) {
  (void)in_sizes; (void)n_in; (void)out_size; (void)ws_size;
  const float* X   = (const float*)d_in[0];
  const float* Hin = (const float*)d_in[1];
  const float* W1  = (const float*)d_in[2];
  const float* b1  = (const float*)d_in[3];
  const float* Wih = (const float*)d_in[4];
  const float* bih = (const float*)d_in[5];
  const float* Whh = (const float*)d_in[6];
  const float* bhh = (const float*)d_in[7];
  const float* W2  = (const float*)d_in[8];
  const float* b2  = (const float*)d_in[9];

  float*  Q    = (float*)d_out;            // [B*A, NA]
  float*  Hout = (float*)d_out + QSZ;      // [B, A, H]
  bf16_t* x1ws = (bf16_t*)d_ws;            // [B*A, H] bf16 (8 MB)

  k_fc1<<<dim3(A_N, B_N / 128, H_N / 64), 256, 0, stream>>>(X, W1, b1, x1ws);
  k_gru<<<dim3(A_N, B_N / 128, H_N / 64), 256, 0, stream>>>(x1ws, Hin, Wih, bih, Whh, bhh, Hout);
  k_fc2<<<dim3(A_N, B_N / 128, 1), 256, 0, stream>>>(Hout, W2, b2, Q);
}
